// MultiHeadAttention_55791625175600
// MI455X (gfx1250) — matrix operand order repaired and checked by reading, not compiled, not run on hardware
//
#include <hip/hip_runtime.h>
#include <math.h>

typedef __attribute__((ext_vector_type(16))) _Float16 v16h;
typedef __attribute__((ext_vector_type(8)))  float    v8f;
typedef __attribute__((ext_vector_type(4)))  float    v4f;

namespace {
constexpr int Bn = 4, Ln = 2048, Hn = 16, Dn = 64;
constexpr int BQ = 128;           // query rows per block (8 waves x 16)
constexpr int WQ = 16;            // query rows per wave
constexpr int KT = 32;            // keys per iteration
constexpr int NWAVES = BQ / WQ;   // 8
constexpr int NTHREADS = NWAVES * 32;  // 256
constexpr int SKPITCH = Dn + 16;  // 160B row stride (32B-aligned, bank-spread)
constexpr int SVPITCH = KT + 16;  // 96B row stride (32B-aligned, bank-spread)
}

// packed f16 max (VOP3P) -- CDNA5 v_pk_max_num_f16
static __device__ inline unsigned pkmax(unsigned a, unsigned b) {
  unsigned d;
  asm("v_pk_max_num_f16 %0, %1, %2" : "=v"(d) : "v"(a), "v"(b));
  return d;
}

__global__ __launch_bounds__(NTHREADS)
void fa_fwd_kernel(const float* __restrict__ Q, const float* __restrict__ K,
                   const float* __restrict__ V, float* __restrict__ O)
{
  __shared__ __align__(32) _Float16 sK[KT * SKPITCH];      // [key][d]  f16
  __shared__ __align__(32) _Float16 sV[Dn * SVPITCH];      // [d][key]  f16 (transposed)
  __shared__ __align__(32) _Float16 sP[NWAVES][WQ * KT];   // per-wave P restage

  const int qtiles = Ln / BQ;
  const int bid = blockIdx.x;
  const int b   = bid / (Hn * qtiles);
  const int rem = bid % (Hn * qtiles);
  const int h   = rem / qtiles;
  const int q0  = (rem % qtiles) * BQ;

  const int tid  = threadIdx.x;
  const int wave = tid >> 5;
  const int lane = tid & 31;
  const int hi   = lane >> 4;   // half-wave
  const int lm   = lane & 15;

  const int qw = q0 + wave * WQ;

  // ---- Q A-fragments: 16x32 f16, two contraction chunks (D=64), scale folded in ----
  v16h qa[2];
  {
    const float* qp = Q + (((size_t)b * Ln + (qw + lm)) * Hn + h) * Dn;
    const float scale = 0.125f;   // 1/sqrt(64)
    #pragma unroll
    for (int c = 0; c < 2; ++c) {
      const int d0 = c * 32 + (hi ? 16 : 0);
      v4f x0 = *(const v4f*)(qp + d0);
      v4f x1 = *(const v4f*)(qp + d0 + 4);
      v4f x2 = *(const v4f*)(qp + d0 + 8);
      v4f x3 = *(const v4f*)(qp + d0 + 12);
      #pragma unroll
      for (int j = 0; j < 4; ++j) {
        qa[c][j]      = (_Float16)(x0[j] * scale);
        qa[c][4 + j]  = (_Float16)(x1[j] * scale);
        qa[c][8 + j]  = (_Float16)(x2[j] * scale);
        qa[c][12 + j] = (_Float16)(x3[j] * scale);
      }
    }
  }

  // ones-column B fragment: column N==0 all-ones over K -> row-sum via WMMA
  v16h vones;
  {
    const _Float16 o = (lm == 0) ? (_Float16)1.0f : (_Float16)0.0f;
    #pragma unroll
    for (int j = 0; j < 16; ++j) vones[j] = o;
  }

  float mrow[8];
  v8f acc[4];   // output accumulators (4 d-chunks)
  v8f accS;     // running softmax normalizer lives in column N=0
  #pragma unroll
  for (int r = 0; r < 8; ++r) { mrow[r] = -__builtin_inff(); accS[r] = 0.f; }
  #pragma unroll
  for (int c = 0; c < 4; ++c)
    #pragma unroll
    for (int r = 0; r < 8; ++r) acc[c][r] = 0.f;

  const int ntiles = (q0 + BQ) / KT;    // causal: keys < q0+BQ
  const int qmax_w = qw + WQ - 1;

  // ---- per-thread staging slots: pointers bumped by constant tile stride ----
  const size_t tstride = (size_t)KT * Hn * Dn;   // elements per key-tile step
  const float* kp[2];
  const float* vp[2];
  int skey[2], sd4[2];
  #pragma unroll
  for (int i = 0; i < 2; ++i) {
    const int idx = tid + i * NTHREADS;          // float4 units, 512 per tile
    skey[i] = idx >> 4;
    sd4[i]  = (idx & 15) << 2;
    const size_t gofs = (((size_t)b * Ln + skey[i]) * Hn + h) * Dn + sd4[i];
    kp[i] = K + gofs;
    vp[i] = V + gofs;
  }

  // prologue: issue loads for tile 0
  v4f kreg[2], vreg[2];
  #pragma unroll
  for (int i = 0; i < 2; ++i) {
    kreg[i] = *(const v4f*)kp[i];
    vreg[i] = *(const v4f*)vp[i];
    kp[i] += tstride; vp[i] += tstride;
  }

  for (int t = 0; t < ntiles; ++t) {
    const int k0 = t * KT;

    __syncthreads();                    // all waves done reading previous tile
    // store staged registers -> LDS (f32 -> f16)
    #pragma unroll
    for (int i = 0; i < 2; ++i) {
      #pragma unroll
      for (int j = 0; j < 4; ++j) {
        sK[skey[i] * SKPITCH + sd4[i] + j]   = (_Float16)kreg[i][j];
        sV[(sd4[i] + j) * SVPITCH + skey[i]] = (_Float16)vreg[i][j];
      }
    }
    __syncthreads();                    // staged tile visible

    // issue next tile's loads now; their wait lands after this tile's compute
    if (t + 1 < ntiles) {
      #pragma unroll
      for (int i = 0; i < 2; ++i) {
        __builtin_prefetch(kp[i] + tstride, 0, 1);   // t+2 toward L2
        __builtin_prefetch(vp[i] + tstride, 0, 1);
        kreg[i] = *(const v4f*)kp[i];
        vreg[i] = *(const v4f*)vp[i];
        kp[i] += tstride; vp[i] += tstride;
      }
    }

    if (k0 <= qmax_w) {                 // wave-uniform; barriers stay outside
      // ---- S = Q K^T : two 16-key subtiles, chained over D=64 ----
      v8f s[2];
      #pragma unroll
      for (int n = 0; n < 2; ++n) {
        const int keyc = n * 16 + lm;   // B layout: lane holds key column N
        v16h kb0 = *(const v16h*)(&sK[keyc * SKPITCH + hi * 16]);
        v16h kb1 = *(const v16h*)(&sK[keyc * SKPITCH + 32 + hi * 16]);
        v8f z;
        #pragma unroll
        for (int r = 0; r < 8; ++r) z[r] = 0.f;
        v8f c0 = __builtin_amdgcn_wmma_f32_16x16x32_f16(false, qa[0], false, kb0,
                                                        (short)0, z, false, false);
        s[n]   = __builtin_amdgcn_wmma_f32_16x16x32_f16(false, qa[1], false, kb1,
                                                        (short)0, c0, false, false);
      }

      // ---- causal mask ----
      const bool needmask = (k0 + KT - 1) > qw;
      if (needmask) {
        #pragma unroll
        for (int r = 0; r < 8; ++r) {
          const int qr = qw + hi * 8 + r;
          #pragma unroll
          for (int n = 0; n < 2; ++n) {
            const int kk = k0 + n * 16 + lm;
            if (kk > qr) s[n][r] = -__builtin_inff();
          }
        }
      }

      // ---- row-max over 16 lanes: packed-f16 butterfly (4 regs carry 8 rows) ----
      unsigned pk[4];
      #pragma unroll
      for (int i = 0; i < 4; ++i) {
        const float a0 = fmaxf(s[0][2 * i],     s[1][2 * i]);
        const float a1 = fmaxf(s[0][2 * i + 1], s[1][2 * i + 1]);
        auto ph = __builtin_amdgcn_cvt_pkrtz(a0, a1);
        pk[i] = __builtin_bit_cast(unsigned, ph);
      }
      #pragma unroll
      for (int i = 0; i < 4; ++i)
        pk[i] = pkmax(pk[i], (unsigned)__builtin_amdgcn_ds_swizzle((int)pk[i], 0x041f));
      #pragma unroll
      for (int i = 0; i < 4; ++i)
        pk[i] = pkmax(pk[i], (unsigned)__builtin_amdgcn_ds_swizzle((int)pk[i], 0x081f));
      #pragma unroll
      for (int i = 0; i < 4; ++i)
        pk[i] = pkmax(pk[i], (unsigned)__builtin_amdgcn_ds_swizzle((int)pk[i], 0x101f));
      #pragma unroll
      for (int i = 0; i < 4; ++i)
        pk[i] = pkmax(pk[i], (unsigned)__builtin_amdgcn_ds_swizzle((int)pk[i], 0x201f));

      // ---- online softmax update + P restage ----
      #pragma unroll
      for (int r = 0; r < 8; ++r) {
        const unsigned u = pk[r >> 1];
        const unsigned short huw = (r & 1) ? (unsigned short)(u >> 16)
                                           : (unsigned short)(u & 0xffff);
        const float mt   = (float)__builtin_bit_cast(_Float16, huw);
        const float mnew = fmaxf(mrow[r], mt);
        const float fac  = __expf(mrow[r] - mnew);
        mrow[r] = mnew;
        #pragma unroll
        for (int n = 0; n < 2; ++n)
          s[n][r] = __expf(s[n][r] - mnew);
        #pragma unroll
        for (int c = 0; c < 4; ++c) acc[c][r] *= fac;
        accS[r] *= fac;                  // normalizer rescales identically
        #pragma unroll
        for (int n = 0; n < 2; ++n)
          sP[wave][(hi * 8 + r) * KT + n * 16 + lm] = (_Float16)s[n][r];
      }

      // ---- load P as A-fragment; acc += P V ; accS += P x ones ----
      v16h pa;
      {
        const _Float16* pr = &sP[wave][lm * KT + (hi ? 16 : 0)];
        #pragma unroll
        for (int j = 0; j < 8; ++j) { pa[j] = pr[j]; pa[8 + j] = pr[8 + j]; }
      }
      #pragma unroll
      for (int c = 0; c < 4; ++c) {
        const int dd = c * 16 + lm;
        v16h vb = *(const v16h*)(&sV[dd * SVPITCH + hi * 16]);
        acc[c] = __builtin_amdgcn_wmma_f32_16x16x32_f16(false, pa, false, vb,
                                                        (short)0, acc[c], false, false);
      }
      accS = __builtin_amdgcn_wmma_f32_16x16x32_f16(false, pa, false, vones,
                                                    (short)0, accS, false, false);
    }
  }

  // ---- epilogue: broadcast row-sum from column 0, normalize, write [B,L,H,D] ----
  #pragma unroll
  for (int r = 0; r < 8; ++r) {
    const int qr = qw + hi * 8 + r;
    const float lsum = __shfl(accS[r], hi * 16, 32);  // N=0 column of this half
    const float inv  = 1.0f / lsum;
    float* op = O + (((size_t)b * Ln + qr) * Hn + h) * Dn;
    #pragma unroll
    for (int c = 0; c < 4; ++c) op[c * 16 + lm] = acc[c][r] * inv;
  }
}

extern "C" void kernel_launch(void* const* d_in, const int* in_sizes, int n_in,
                              void* d_out, int out_size, void* d_ws, size_t ws_size,
                              hipStream_t stream) {
  const float* Q = (const float*)d_in[0];
  const float* K = (const float*)d_in[1];
  const float* V = (const float*)d_in[2];
  float* O = (float*)d_out;
  dim3 grid(Bn * Hn * (Ln / BQ));   // 1024 blocks: (b, h, q-tile)
  dim3 block(NTHREADS);             // 8 waves
  hipLaunchKernelGGL(fa_fwd_kernel, grid, block, 0, stream, Q, K, V, O);
}
